// LengthRegulator_64278480552087
// MI455X (gfx1250) — matrix operand order repaired and checked by reading, not compiled, not run on hardware
//
#include <hip/hip_runtime.h>

// ---------------------------------------------------------------------------
// LengthRegulator for MI455X (gfx1250): conv-duration-predictor via bf16 WMMA
// with TDM (tensor_load_to_lds) weight staging; expand-by-durations via
// coalesced float4 gather.
// ---------------------------------------------------------------------------

typedef __attribute__((ext_vector_type(16))) __bf16       v16bf;
typedef __attribute__((ext_vector_type(8)))  float        v8f;
typedef __attribute__((ext_vector_type(4)))  unsigned int v4u;
typedef __attribute__((ext_vector_type(8)))  unsigned int v8u;

#define B_    32
#define T_    256
#define C_    512
#define H_    512
#define M_    (B_ * T_)        // 8192 tokens
#define TOUT_ 2048             // 8 * T
#define RES_ELEMS ((size_t)B_ * TOUT_ * C_)

static __device__ inline unsigned pk_bf16(float lo, float hi) {
  union { __bf16 h[2]; unsigned u; } z;
  z.h[0] = (__bf16)lo; z.h[1] = (__bf16)hi;
  return z.u;
}

// ---------------------------------------------------------------------------
// Weight prep: W[3][cin][cout] f32  ->  Wt[3][cout][cin] bf16
// (B-matrix fragments and the TDM tile want K (=cin) contiguous per column)
// ---------------------------------------------------------------------------
__global__ void lr_wtrans_kernel(const float* __restrict__ W,
                                 __bf16* __restrict__ Wt) {
  int i = blockIdx.x * blockDim.x + threadIdx.x;   // over 3*512*512
  if (i >= 3 * C_ * H_) return;
  int cin  = i & (C_ - 1);
  int cout = (i >> 9) & (H_ - 1);
  int k    = i >> 18;
  Wt[i] = (__bf16)W[(k * C_ + cin) * H_ + cout];
}

// ---------------------------------------------------------------------------
// Fused conv1d(k=3, same) + bias + LayerNorm + ReLU.
// One 256-thread block (8 waves) computes 16 token-rows x 512 channels.
// Wave w owns columns [w*64, w*64+64): 4 C/D fragments, f32 accumulation.
// Per tap: stage the full 16x512 bf16 A panel once (1 barrier pair), then run
// 16 barrier-free K-steps of v_wmma_f32_16x16x32_bf16. B slabs (64 couts x
// 32 cin bf16 = 4 KB) are DMA'd per wave by the Tensor Data Mover
// (tensor_load_to_lds, TENSORcnt). A dtype is a compile-time template.
// ---------------------------------------------------------------------------
template <bool A_IS_F32>
__global__ __launch_bounds__(256)
void lr_conv_ln_relu_kernel(const float*  __restrict__ Af32,
                            const __bf16* __restrict__ Abf,
                            const __bf16* __restrict__ Wt,   // [3][cout][cin] bf16
                            const float*  __restrict__ bias,
                            const float*  __restrict__ gamma,
                            const float*  __restrict__ beta,
                            __bf16* __restrict__ out_bf,
                            float*  __restrict__ out_f32) {
  __shared__ __bf16 Alds[16][C_];          // 16 KB : full per-tap A panel
  __shared__ union {                       // 32 KB, phases do not overlap
    __bf16 Bt[512][32];                    //   K-loop:  per-wave B slabs
    float  Y[16][512];                     //   epilog:  pre-LN activations
  } uS;
  __shared__ float red_s[16][16];
  __shared__ float red_q[16][16];
  __shared__ float mu_s[16], rs_s[16];

  const int tid   = threadIdx.x;
  const int lane  = tid & 31;              // wave32
  const int wv    = tid >> 5;              // 0..7
  const int m0    = blockIdx.x * 16;       // first token row of this tile
  const int wbase = wv * 64;               // columns owned by this wave

  // -------- TDM descriptor pieces (wave-uniform scalars, hoisted) --------
  // Group1 is fully constant: workgroup_mask=0, data_size=1 (2B),
  // tensor_dim0=512, tensor_dim1=1536, tile_dim0=32, tile_dim1=64,
  // tile_dim2=0, tensor_dim0_stride=512, tensor_dim1_stride=0.
  const v8u g1 = {1u << 16,   512u << 16, 1536u << 16, 32u << 16,
                  64u,        512u,       0u,          0u};
  const unsigned lds_b = __builtin_amdgcn_readfirstlane(
      (unsigned)(unsigned long long)(void*)&uS.Bt[wbase][0]);
  const unsigned long long ga_w =
      (unsigned long long)Wt + (size_t)wbase * C_ * 2;     // wave's W base
  const unsigned long long ga_base =
      ((unsigned long long)__builtin_amdgcn_readfirstlane(
           (unsigned)(ga_w >> 32)) << 32) |
      (unsigned long long)__builtin_amdgcn_readfirstlane((unsigned)ga_w);

  v8f acc[4];
#pragma unroll
  for (int f = 0; f < 4; ++f) acc[f] = v8f{0, 0, 0, 0, 0, 0, 0, 0};

  for (int tap = 0; tap < 3; ++tap) {
    __syncthreads();                       // all waves done with prev A panel

    // ---- stage full A panel: 16 rows x 512 cin, tap-shifted, vectorized.
    // Thread -> (row r = tid>>4, 32-elem segment sg = tid&15).
    {
      int r = tid >> 4, sg = tid & 15;
      int m  = m0 + r;
      int t  = m & (T_ - 1);
      int ts = t + tap - 1;                // 'same' padding
      bool ok = (unsigned)ts < (unsigned)T_;
      int  msrc = ok ? (m + tap - 1) : m;  // clamp to a safe row
      if (A_IS_F32) {
        const float4* src =
            (const float4*)(Af32 + (size_t)msrc * C_ + sg * 32);
#pragma unroll
        for (int j = 0; j < 4; ++j) {      // 8 floats -> 8 bf16 per step
          float4 a = src[2 * j], b = src[2 * j + 1];
          uint4 o;
          o.x = pk_bf16(a.x, a.y); o.y = pk_bf16(a.z, a.w);
          o.z = pk_bf16(b.x, b.y); o.w = pk_bf16(b.z, b.w);
          if (!ok) { o.x = 0; o.y = 0; o.z = 0; o.w = 0; }
          *(uint4*)&Alds[r][sg * 32 + j * 8] = o;
        }
      } else {
        const uint4* src =
            (const uint4*)(Abf + (size_t)msrc * C_ + sg * 32);
#pragma unroll
        for (int j = 0; j < 4; ++j) {      // 8 bf16 per step
          uint4 o = src[j];
          if (!ok) { o.x = 0; o.y = 0; o.z = 0; o.w = 0; }
          *(uint4*)&Alds[r][sg * 32 + j * 8] = o;
        }
      }
    }
    __syncthreads();                       // A panel visible to the block

    // ---- 16 barrier-free K-steps over this tap
    for (int kb = 0; kb < C_ / 32; ++kb) {
      // issue TDM: DMA this wave's 64x32 bf16 weight slab into LDS.
      // (safe single-buffer reuse: previous step's s_wait_dscnt-before-WMMA
      //  drained the B-fragment reads before this issues)
      {
        unsigned long long ga =
            ga_base + ((size_t)tap * H_ * C_ + (size_t)kb * 32) * 2;
        v4u g0;
        g0.x = 1u;                                     // count=1 (valid D#)
        g0.y = lds_b;                                  // lds_addr
        g0.z = (unsigned)ga;                           // global_addr[31:0]
        g0.w = ((unsigned)(ga >> 32) & 0x01ffffffu) |
               0x80000000u;                            // addr[56:32] | type=2
        asm volatile("tensor_load_to_lds %0, %1" : : "s"(g0), "s"(g1)
                     : "memory");
      }

      // A fragment per ISA 16-bit 16x32 layout (two ds_load_b128)
      v16bf afrag;
      {
        int ar = lane & 15;
        int kh = (lane < 16) ? 0 : 8;
        const __bf16* p = &Alds[ar][kb * 32];
#pragma unroll
        for (int i = 0; i < 8; ++i) {
          afrag[i]     = p[kh + i];            // K = kh .. kh+7     (VGPR 0-3)
          afrag[8 + i] = p[kh + 16 + i];       // K = kh+16 .. kh+23 (VGPR 4-7)
        }
      }

      __builtin_amdgcn_s_wait_tensorcnt(0);    // this wave's B slab landed

      // 4 B fragments + WMMA
#pragma unroll
      for (int f = 0; f < 4; ++f) {
        v16bf bfrag;
        int col = wbase + f * 16 + (lane & 15);
        int ko  = (lane < 16) ? 0 : 8;
        const __bf16* p = &uS.Bt[col][ko];
#pragma unroll
        for (int i = 0; i < 8; ++i) { bfrag[i] = p[i]; bfrag[8 + i] = p[16 + i]; }
        acc[f] = __builtin_amdgcn_wmma_f32_16x16x32_bf16(
            false, afrag, false, bfrag, (short)0, acc[f], false, false);
      }
    }
  }
  __syncthreads();                         // all waves done reading B slabs

  // ---- epilogue: scatter D fragments (+bias) to LDS, then LayerNorm+ReLU
#pragma unroll
  for (int f = 0; f < 4; ++f) {
    int col   = wbase + f * 16 + (lane & 15);
    int rbase = (lane < 16) ? 0 : 8;
    float bsv = bias[col];
#pragma unroll
    for (int v = 0; v < 8; ++v) uS.Y[rbase + v][col] = acc[f][v] + bsv;
  }
  __syncthreads();

  int r = tid >> 4, sg = tid & 15;             // 16 threads per row
  float s = 0.0f, q = 0.0f;
  for (int j = 0; j < 32; ++j) {
    float v = uS.Y[r][sg * 32 + j];
    s += v; q += v * v;
  }
  red_s[r][sg] = s; red_q[r][sg] = q;
  __syncthreads();
  if (sg == 0) {
    float ts = 0.0f, tq = 0.0f;
    for (int j = 0; j < 16; ++j) { ts += red_s[r][j]; tq += red_q[r][j]; }
    float mu  = ts * (1.0f / H_);
    float var = tq * (1.0f / H_) - mu * mu;
    mu_s[r] = mu;
    rs_s[r] = rsqrtf(var + 1e-5f);
  }
  __syncthreads();
  float mu = mu_s[r], rs = rs_s[r];
  int m = m0 + r;
  for (int j = 0; j < 32; ++j) {
    int c = sg * 32 + j;
    float v = (uS.Y[r][c] - mu) * rs * gamma[c] + beta[c];
    v = v > 0.0f ? v : 0.0f;
    if (out_bf)  out_bf[(size_t)m * H_ + c]  = (__bf16)v;
    if (out_f32) out_f32[(size_t)m * H_ + c] = v;
  }
}

// ---------------------------------------------------------------------------
// Linear H->1: one wave per token, shuffle reduction. ALPHA == 1.0.
// ---------------------------------------------------------------------------
__global__ __launch_bounds__(256)
void lr_linear_kernel(const float* __restrict__ h,
                      const float* __restrict__ Wl,
                      const float* __restrict__ bl,
                      float* __restrict__ ll) {
  int m    = blockIdx.x * 8 + (threadIdx.x >> 5);
  int lane = threadIdx.x & 31;
  float s = 0.0f;
#pragma unroll
  for (int i = 0; i < H_ / 32; ++i) {
    int c = lane + i * 32;
    s += h[(size_t)m * H_ + c] * Wl[c];
  }
#pragma unroll
  for (int o = 16; o; o >>= 1) s += __shfl_down(s, o, 32);
  if (lane == 0) ll[m] = s + bl[0];
}

// ---------------------------------------------------------------------------
// Per-batch inclusive cumsum of durations (T=256, one block per batch).
// ---------------------------------------------------------------------------
__global__ void lr_cumsum_kernel(const int* __restrict__ dur,
                                 int* __restrict__ csum) {
  __shared__ int s[T_];
  int b = blockIdx.x, t = threadIdx.x;
  s[t] = dur[b * T_ + t];
  __syncthreads();
  int acc = 0;
  for (int i = 0; i <= t; ++i) acc += s[i];
  csum[b * T_ + t] = acc;
}

// ---------------------------------------------------------------------------
// Expand: one block per (b, t_out); searchsorted(csum, t_out, 'right'),
// then copy/zero 512 floats as 128 x float4 (fully coalesced; this is the
// HBM-bound part: ~134 MB written + ~134 MB gathered -> ~12 us at 23.3 TB/s).
// ---------------------------------------------------------------------------
__global__ __launch_bounds__(128)
void lr_expand_kernel(const float* __restrict__ x,
                      const int* __restrict__ csum,
                      float* __restrict__ res) {
  int b = blockIdx.x >> 11;                // / 2048
  int t = blockIdx.x & (TOUT_ - 1);
  __shared__ int sIdx, sValid;
  if (threadIdx.x == 0) {
    const int* c = csum + b * T_;
    int lo = 0, hi = T_;                   // first i with c[i] > t
    while (lo < hi) {
      int mid = (lo + hi) >> 1;
      if (c[mid] > t) hi = mid; else lo = mid + 1;
    }
    sValid = (t < c[T_ - 1]) ? 1 : 0;
    sIdx   = (lo < T_) ? lo : (T_ - 1);    // clip
  }
  __syncthreads();
  int idx = sIdx, valid = sValid;
  const float4* src = (const float4*)(x + ((size_t)b * T_ + idx) * C_);
  float4*       dst = (float4*)(res + ((size_t)b * TOUT_ + t) * C_);
  float4 v = valid ? src[threadIdx.x] : make_float4(0.f, 0.f, 0.f, 0.f);
  dst[threadIdx.x] = v;
}

// ---------------------------------------------------------------------------
extern "C" void kernel_launch(void* const* d_in, const int* in_sizes, int n_in,
                              void* d_out, int out_size, void* d_ws,
                              size_t ws_size, hipStream_t stream) {
  (void)in_sizes; (void)n_in; (void)out_size; (void)ws_size;

  const float* x   = (const float*)d_in[0];
  const int*   dur = (const int*)d_in[1];
  const float* W1  = (const float*)d_in[2];
  const float* b1  = (const float*)d_in[3];
  const float* g1  = (const float*)d_in[4];
  const float* be1 = (const float*)d_in[5];
  const float* W2  = (const float*)d_in[6];
  const float* b2  = (const float*)d_in[7];
  const float* g2  = (const float*)d_in[8];
  const float* be2 = (const float*)d_in[9];
  const float* Wl  = (const float*)d_in[10];
  const float* bl  = (const float*)d_in[11];
  float* out = (float*)d_out;

  // Workspace layout (all offsets 16B-aligned):
  //  [0)       W1bf : 3*512*512 bf16 = 1.5 MB
  //  [+1.5MB)  W2bf : 1.5 MB
  //  [+3MB)    H1bf : 8192*512 bf16 = 8 MB   (layer-1 out, layer-2 A)
  //  [+11MB)   H2f  : 8192*512 f32  = 16 MB  (layer-2 out, linear in)
  //  [+27MB)   csum : 32*256 i32
  char* ws = (char*)d_ws;
  const size_t W_BYTES  = (size_t)3 * C_ * H_ * 2;         // 1,572,864
  const size_t H1_BYTES = (size_t)M_ * H_ * 2;             // 8,388,608
  const size_t H2_BYTES = (size_t)M_ * H_ * 4;             // 16,777,216
  __bf16* W1bf = (__bf16*)ws;
  __bf16* W2bf = (__bf16*)(ws + W_BYTES);
  __bf16* H1bf = (__bf16*)(ws + 2 * W_BYTES);
  float*  H2f  = (float*)(ws + 2 * W_BYTES + H1_BYTES);
  int*    csum = (int*)(ws + 2 * W_BYTES + H1_BYTES + H2_BYTES);

  const int WTOT = 3 * C_ * H_;
  lr_wtrans_kernel<<<(WTOT + 255) / 256, 256, 0, stream>>>(W1, W1bf);
  lr_wtrans_kernel<<<(WTOT + 255) / 256, 256, 0, stream>>>(W2, W2bf);

  // Layer 1: f32 x in, bf16 hidden out.
  lr_conv_ln_relu_kernel<true><<<M_ / 16, 256, 0, stream>>>(
      x, nullptr, W1bf, b1, g1, be1, H1bf, nullptr);
  // Layer 2: bf16 hidden in, f32 out for the projection.
  lr_conv_ln_relu_kernel<false><<<M_ / 16, 256, 0, stream>>>(
      nullptr, H1bf, W2bf, b2, g2, be2, nullptr, H2f);

  // log_lengths -> d_out tail
  lr_linear_kernel<<<M_ / 8, 256, 0, stream>>>(H2f, Wl, bl, out + RES_ELEMS);

  // expand-by-durations -> d_out head
  lr_cumsum_kernel<<<B_, T_, 0, stream>>>(dur, csum);
  lr_expand_kernel<<<B_ * TOUT_, 128, 0, stream>>>(x, csum, out);
}
